// isGNN_65549790871884
// MI455X (gfx1250) — matrix operand order repaired and checked by reading, not compiled, not run on hardware
//
#include <hip/hip_runtime.h>

typedef __bf16 bf16_t;
typedef bf16_t v16bf __attribute__((ext_vector_type(16)));
typedef float v8f __attribute__((ext_vector_type(8)));

#define NEG_SLOPE 0.2f

// ---------- helpers ----------
__device__ __forceinline__ unsigned short f32_to_bf16(float f) {
  unsigned u = __float_as_uint(f);
  unsigned r = u + 0x7FFFu + ((u >> 16) & 1u);  // round-to-nearest-even
  return (unsigned short)(r >> 16);
}

// monotone float -> uint key (unsigned order == float order), for atomicMax
__device__ __forceinline__ unsigned fkey(float f) {
  unsigned u = __float_as_uint(f);
  return (u & 0x80000000u) ? ~u : (u | 0x80000000u);
}
__device__ __forceinline__ float fkey_inv(unsigned k) {
  unsigned u = (k & 0x80000000u) ? (k & 0x7FFFFFFFu) : ~k;
  return __uint_as_float(u);
}

// ---------- f32 -> bf16 convert (vectorized x4) ----------
__global__ void k_cvt_bf16_x4(const float* __restrict__ in,
                              unsigned short* __restrict__ out, long n4) {
  long i = (long)blockIdx.x * blockDim.x + threadIdx.x;
  long stride = (long)gridDim.x * blockDim.x;
  for (; i < n4; i += stride) {
    float4 v = ((const float4*)in)[i];
    ushort4 o;
    o.x = f32_to_bf16(v.x); o.y = f32_to_bf16(v.y);
    o.z = f32_to_bf16(v.z); o.w = f32_to_bf16(v.w);
    ((ushort4*)out)[i] = o;
  }
}

// ---------- fused dual WMMA GEMM ----------
// XL[N,256] = A[N,K](bf16) * Wl[K,256](bf16) + bl
// XR[N,256] = A[N,K](bf16) * Wr[K,256](bf16) + br
// A tile staged once, reused for both projections; double-buffered LDS,
// global loads register-staged during WMMA compute -> 1 barrier / K-step.
#define BM 128
#define BN 64
#define BK 32
#define LDSTR 40  // ushort stride: 80B rows -> 16B-aligned uint4 stores

__global__ __launch_bounds__(256) void k_gemm_dual(
    const unsigned short* __restrict__ A,
    const unsigned short* __restrict__ Wl, const unsigned short* __restrict__ Wr,
    const float* __restrict__ bl, const float* __restrict__ br,
    float* __restrict__ XL, float* __restrict__ XR, int Nrows, int K) {
  __shared__ unsigned short sA[2][BM][LDSTR];    // [buf][row][k]
  __shared__ unsigned short sBl[2][BN][LDSTR];   // [buf][col][k] (transposed)
  __shared__ unsigned short sBr[2][BN][LDSTR];
  const int tid = threadIdx.x;
  const int wid = tid >> 5;
  const int lane = tid & 31;
  const int hlf = lane >> 4;
  const int l16 = lane & 15;
  const int rowBlock = blockIdx.x * BM;
  const int colBlock = blockIdx.y * BN;

  // staging registers for the next K tile
  uint4 aR0, aR1, blR, brR;
  const int rA0 = tid >> 2, cA0 = (tid & 3) * 8;
  const int rA1 = (tid + 256) >> 2, cA1 = ((tid + 256) & 3) * 8;
  const int kB = tid >> 3, cB = (tid & 7) * 8;

  auto loadRegs = [&](int k0) {
    int gr0 = rowBlock + rA0; if (gr0 >= Nrows) gr0 = Nrows - 1;
    int gr1 = rowBlock + rA1; if (gr1 >= Nrows) gr1 = Nrows - 1;
    aR0 = *(const uint4*)(A + (long)gr0 * K + k0 + cA0);
    aR1 = *(const uint4*)(A + (long)gr1 * K + k0 + cA1);
    blR = *(const uint4*)(Wl + (long)(k0 + kB) * 256 + colBlock + cB);
    brR = *(const uint4*)(Wr + (long)(k0 + kB) * 256 + colBlock + cB);
  };
  auto storeLds = [&](int buf) {
    *(uint4*)&sA[buf][rA0][cA0] = aR0;
    *(uint4*)&sA[buf][rA1][cA1] = aR1;
    unsigned pl[4] = {blR.x, blR.y, blR.z, blR.w};
    unsigned pr[4] = {brR.x, brR.y, brR.z, brR.w};
#pragma unroll
    for (int j = 0; j < 4; ++j) {
      sBl[buf][cB + 2 * j][kB]     = (unsigned short)(pl[j] & 0xFFFFu);
      sBl[buf][cB + 2 * j + 1][kB] = (unsigned short)(pl[j] >> 16);
      sBr[buf][cB + 2 * j][kB]     = (unsigned short)(pr[j] & 0xFFFFu);
      sBr[buf][cB + 2 * j + 1][kB] = (unsigned short)(pr[j] >> 16);
    }
  };

  v8f accL[4], accR[4];
#pragma unroll
  for (int i = 0; i < 4; ++i) {
    accL[i] = (v8f){0.f,0.f,0.f,0.f,0.f,0.f,0.f,0.f};
    accR[i] = (v8f){0.f,0.f,0.f,0.f,0.f,0.f,0.f,0.f};
  }

  loadRegs(0);
  storeLds(0);
  __syncthreads();

  int cur = 0;
  for (int k0 = 0; k0 < K; k0 += BK) {
    const bool more = (k0 + BK) < K;
    if (more) loadRegs(k0 + BK);
    if (k0 + 2 * BK < K) {   // keep a speculative global_prefetch_b8 ahead
      __builtin_prefetch(A + (long)(rowBlock + (tid >> 1)) * K + k0 + 2 * BK, 0, 0);
    }

    // A fragment (this code's contiguous k order: VGPRv -> K = 16*half + 2*v)
    union { v16bf v; unsigned u[8]; } fa;
#pragma unroll
    for (int vv = 0; vv < 8; ++vv) {
      int kk = 16 * hlf + 2 * vv;
      fa.u[vv] = *(const unsigned*)&sA[cur][wid * 16 + l16][kk];
    }
    // Wl side
#pragma unroll
    for (int nb = 0; nb < 4; ++nb) {
      union { v16bf v; unsigned u[8]; } fb;
#pragma unroll
      for (int vv = 0; vv < 8; ++vv) {
        int kk = 16 * hlf + 2 * vv;   // same contiguous k order as A
        fb.u[vv] = *(const unsigned*)&sBl[cur][nb * 16 + l16][kk];
      }
      accL[nb] = __builtin_amdgcn_wmma_f32_16x16x32_bf16(
          false, fa.v, false, fb.v, (short)0, accL[nb], false, false);
    }
    // Wr side (reuses fa)
#pragma unroll
    for (int nb = 0; nb < 4; ++nb) {
      union { v16bf v; unsigned u[8]; } fb;
#pragma unroll
      for (int vv = 0; vv < 8; ++vv) {
        int kk = 16 * hlf + 2 * vv;
        fb.u[vv] = *(const unsigned*)&sBr[cur][nb * 16 + l16][kk];
      }
      accR[nb] = __builtin_amdgcn_wmma_f32_16x16x32_bf16(
          false, fa.v, false, fb.v, (short)0, accR[nb], false, false);
    }

    if (more) storeLds(cur ^ 1);
    __syncthreads();
    cur ^= 1;
  }

  // D layout: VGPR i -> M = i + 8*half, N = lane%16
#pragma unroll
  for (int nb = 0; nb < 4; ++nb) {
#pragma unroll
    for (int i = 0; i < 8; ++i) {
      int r = rowBlock + wid * 16 + hlf * 8 + i;
      int col = colBlock + nb * 16 + l16;
      if (r < Nrows) {
        XL[(long)r * 256 + col] = accL[nb][i] + bl[col];
        XR[(long)r * 256 + col] = accR[nb][i] + br[col];
      }
    }
  }
}

// ---------- init ----------
__global__ void k_init_out(float* __restrict__ hout, const float* __restrict__ bias,
                           long total) {
  long i = (long)blockIdx.x * blockDim.x + threadIdx.x;
  long stride = (long)gridDim.x * blockDim.x;
  for (; i < total; i += stride) hout[i] = bias[i & 255];
}
__global__ void k_reset(unsigned* __restrict__ nmax, float* __restrict__ nsum, int total) {
  int i = blockIdx.x * blockDim.x + threadIdx.x;
  if (i < total) { nmax[i] = 0u; nsum[i] = 0.f; }
}

// ---------- edge pass 1: logits + segment max ----------
__global__ __launch_bounds__(256) void k_edge_logits(
    const float* __restrict__ XL, const float* __restrict__ XR,
    const int* __restrict__ eSrc, const int* __restrict__ eDst,
    const float* __restrict__ att, float* __restrict__ logits,
    unsigned* __restrict__ nmax, int E, int Nn) {
  int widx = (blockIdx.x * blockDim.x + threadIdx.x) >> 5;  // one wave per edge
  int lane = threadIdx.x & 31;
  if (widx >= E + Nn) return;
  int s, d;
  if (widx < E) { s = eSrc[widx]; d = eDst[widx]; } else { s = widx - E; d = s; }
  const float* xl = XL + (long)s * 256;
  const float* xr = XR + (long)d * 256;
#pragma unroll
  for (int h = 0; h < 4; ++h) {
    float acc = 0.f;
#pragma unroll
    for (int j = 0; j < 2; ++j) {
      int c = h * 64 + lane + 32 * j;
      float v = xl[c] + xr[c];
      v = (v > 0.f) ? v : v * NEG_SLOPE;       // leaky_relu
      acc += att[c] * v;                        // att flattened [H*Ch]
    }
#pragma unroll
    for (int off = 16; off > 0; off >>= 1) acc += __shfl_xor(acc, off);
    if (lane == 0) {
      logits[(long)widx * 4 + h] = acc;
      atomicMax(&nmax[d * 4 + h], fkey(acc));
    }
  }
}

// ---------- edge pass 2: exp + segment sum ----------
__global__ void k_edge_exp(float* __restrict__ logits, const int* __restrict__ eDst,
                           const unsigned* __restrict__ nmax, float* __restrict__ nsum,
                           int E, int Nn) {
  long i = (long)blockIdx.x * blockDim.x + threadIdx.x;
  long total = (long)(E + Nn) * 4;
  if (i >= total) return;
  int e = (int)(i >> 2), h = (int)(i & 3);
  int d = (e < E) ? eDst[e] : (e - E);
  float m = fkey_inv(nmax[d * 4 + h]);
  float v = __expf(logits[i] - m);
  logits[i] = v;                                // overwrite with exp value
  atomicAdd(&nsum[d * 4 + h], v);
}

// ---------- edge pass 3: normalized message scatter ----------
__global__ __launch_bounds__(256) void k_edge_scatter(
    const float* __restrict__ XL, const float* __restrict__ logits,
    const float* __restrict__ nsum,
    const int* __restrict__ eSrc, const int* __restrict__ eDst,
    float* __restrict__ hout, int E, int Nn) {
  int widx = (blockIdx.x * blockDim.x + threadIdx.x) >> 5;
  int lane = threadIdx.x & 31;
  if (widx >= E + Nn) return;
  int s, d;
  if (widx < E) { s = eSrc[widx]; d = eDst[widx]; } else { s = widx - E; d = s; }
  const float* xl = XL + (long)s * 256;
  float al[4];
#pragma unroll
  for (int h = 0; h < 4; ++h) al[h] = logits[(long)widx * 4 + h] / nsum[d * 4 + h];
  float* od = hout + (long)d * 256;
#pragma unroll
  for (int j = 0; j < 8; ++j) {               // c = lane + 32*j  =>  head = j>>1 (const)
    int c = lane + 32 * j;
    atomicAdd(&od[c], xl[c] * al[j >> 1]);
  }
}

// ---------- relu ----------
__global__ void k_relu(float* __restrict__ h, long n) {
  long i = (long)blockIdx.x * blockDim.x + threadIdx.x;
  long stride = (long)gridDim.x * blockDim.x;
  for (; i < n; i += stride) { float v = h[i]; h[i] = v > 0.f ? v : 0.f; }
}

// ---------- ragged segment-mean pooling + modality concat ----------
__global__ __launch_bounds__(256) void k_pool(
    const float* __restrict__ h, const int* __restrict__ istns,
    const int* __restrict__ nsV, const int* __restrict__ nsA,
    const int* __restrict__ nsT, const int* __restrict__ nModP,
    float* __restrict__ out, int S) {
  int s = blockIdx.x, m = blockIdx.y, c = threadIdx.x;
  int nMod = *nModP;
  int act0 = (*nsV > 0), act1 = (*nsA > 0), act2 = (*nsT > 0);
  int actm = (m == 0) ? act0 : (m == 1 ? act1 : act2);
  long oidx = (long)s * 768 + m * 256 + c;
  if (!actm) { out[oidx] = 0.f; return; }
  int idx = 0;
  if (m >= 1) idx += act0;
  if (m >= 2) idx += act1;
  int b0 = istns[s], b1 = istns[s + 1];
  float sum = 0.f;
  for (int r = b0; r < b1; ++r) sum += h[((long)idx * nMod + r) * 256 + c];
  out[oidx] = sum / (float)(b1 - b0);
}

// ---------- host launcher ----------
extern "C" void kernel_launch(void* const* d_in, const int* in_sizes, int n_in,
                              void* d_out, int out_size, void* d_ws, size_t ws_size,
                              hipStream_t stream) {
  (void)n_in; (void)out_size; (void)ws_size;
  const float* x = (const float*)d_in[0];
  const int* eSrc = (const int*)d_in[1];
  int E = in_sizes[1] / 2;
  const int* eDst = eSrc + E;
  const int* nsV = (const int*)d_in[2];
  const int* nsA = (const int*)d_in[3];
  const int* nsT = (const int*)d_in[4];
  const int* nModP = (const int*)d_in[5];
  const int* istns = (const int*)d_in[6];
  int S = in_sizes[6] - 1;
  int IN0 = in_sizes[7] / 256;   // layer-0 din (512), from Wl size
  int N = in_sizes[0] / IN0;     // 60000 nodes

  // bump-allocate scratch (~257 MB)
  char* wsp = (char*)d_ws;
  auto alloc = [&](size_t b) -> void* {
    void* p = (void*)wsp;
    wsp += (b + 255) & ~(size_t)255;
    return p;
  };
  unsigned short* hb  = (unsigned short*)alloc((size_t)N * IN0 * 2);
  float* hout         = (float*)alloc((size_t)N * 256 * 4);
  float* XL           = (float*)alloc((size_t)N * 256 * 4);
  float* XR           = (float*)alloc((size_t)N * 256 * 4);
  unsigned short* wlb = (unsigned short*)alloc((size_t)IN0 * 256 * 2);
  unsigned short* wrb = (unsigned short*)alloc((size_t)IN0 * 256 * 2);
  float* logits       = (float*)alloc((size_t)(E + N) * 4 * 4);
  unsigned* nmax      = (unsigned*)alloc((size_t)N * 4 * 4);
  float* nsum         = (float*)alloc((size_t)N * 4 * 4);

  const float* hin = x;
  int EW = E + N;
  for (int li = 0; li < 4; ++li) {
    int din = (li == 0) ? IN0 : 256;
    int pb = 7 + li * 6;
    const float* Wl   = (const float*)d_in[pb + 0];
    const float* bl   = (const float*)d_in[pb + 1];
    const float* Wr   = (const float*)d_in[pb + 2];
    const float* br   = (const float*)d_in[pb + 3];
    const float* att  = (const float*)d_in[pb + 4];
    const float* bias = (const float*)d_in[pb + 5];

    long nA4 = (long)N * din / 4;
    k_cvt_bf16_x4<<<dim3((unsigned)((nA4 + 255) / 256)), dim3(256), 0, stream>>>(hin, hb, nA4);
    long nW4 = (long)din * 256 / 4;
    k_cvt_bf16_x4<<<dim3((unsigned)((nW4 + 255) / 256)), dim3(256), 0, stream>>>(Wl, wlb, nW4);
    k_cvt_bf16_x4<<<dim3((unsigned)((nW4 + 255) / 256)), dim3(256), 0, stream>>>(Wr, wrb, nW4);

    dim3 gg((unsigned)((N + BM - 1) / BM), 4);
    k_gemm_dual<<<gg, 256, 0, stream>>>(hb, wlb, wrb, bl, br, XL, XR, N, din);

    long tot = (long)N * 256;
    k_init_out<<<dim3((unsigned)((tot + 255) / 256)), 256, 0, stream>>>(hout, bias, tot);
    k_reset<<<dim3((unsigned)((N * 4 + 255) / 256)), 256, 0, stream>>>(nmax, nsum, N * 4);

    unsigned ewb = (unsigned)(((long)EW * 32 + 255) / 256);
    k_edge_logits<<<dim3(ewb), 256, 0, stream>>>(XL, XR, eSrc, eDst, att, logits, nmax, E, N);
    k_edge_exp<<<dim3((unsigned)(((long)EW * 4 + 255) / 256)), 256, 0, stream>>>(logits, eDst, nmax, nsum, E, N);
    k_edge_scatter<<<dim3(ewb), 256, 0, stream>>>(XL, logits, nsum, eSrc, eDst, hout, E, N);
    k_relu<<<dim3((unsigned)((tot + 255) / 256)), 256, 0, stream>>>(hout, tot);
    hin = hout;
  }

  k_pool<<<dim3((unsigned)S, 3), 256, 0, stream>>>(hout, istns, nsV, nsA, nsT, nModP,
                                                   (float*)d_out, S);
}
